// TrfEdgeNetL1_79645873537754
// MI455X (gfx1250) — matrix operand order repaired and checked by reading, not compiled, not run on hardware
//
#include <hip/hip_runtime.h>

typedef _Float16 f16;
typedef _Float16 v16h __attribute__((ext_vector_type(16)));
typedef float    v8f  __attribute__((ext_vector_type(8)));

#define N_NODES 50000
#define N_EDGES 400000
#define F_IN    128
#define F_E     32
#define H_HEADS 4
#define C_CH    64
#define HC      256          // H*C
#define NCLASS  16
#define NCOLS   832          // 256 q | 256 k | 256 v | 64 skip
#define MTILES  (N_NODES/16) // 3125 (exact)
#define NTILES  (NCOLS/16)   // 52
#define KTILES  (F_IN/32)    // 4
#define BN_EPS  1e-5f

// ---------------------------------------------------------------------------
// Pack x (f32 row-major) into WMMA A-fragment layout (16-bit A 16x32, ISA 7.12.2)
// apack[((mt*4+ks)*32+lane)*8+p] = halves {A[m][k], A[m][k+1]}
// ---------------------------------------------------------------------------
__global__ void pack_a_kernel(const float* __restrict__ x, unsigned int* __restrict__ apack) {
    int t = blockIdx.x * blockDim.x + threadIdx.x;
    const int total = MTILES * KTILES * 32 * 8;   // 3,200,000
    if (t >= total) return;
    int p    = t & 7;
    int lane = (t >> 3) & 31;
    int ks   = (t >> 8) & 3;
    int mt   = t >> 10;
    int m  = mt * 16 + (lane & 15);
    int kb = (lane < 16) ? 0 : 8;
    int kk = (p < 4) ? (kb + 2 * p) : (kb + 8 + 2 * p);
    int k  = ks * 32 + kk;
    const float* row = x + (size_t)m * F_IN + k;
    union { f16 h[2]; unsigned int u; } cv;
    cv.h[0] = (f16)row[0];
    cv.h[1] = (f16)row[1];
    apack[t] = cv.u;
}

// ---------------------------------------------------------------------------
// Pack [Wq|Wk|Wv|Wskip] (f32, K=128 x 832) into B-fragment layout + bias vector
// bpack[((ks*52+nt)*32+lane)*8+v] = halves {B[kB][col], B[kB+1][col]}
// ---------------------------------------------------------------------------
__global__ void pack_b_kernel(const float* __restrict__ Wq, const float* __restrict__ Wk,
                              const float* __restrict__ Wv, const float* __restrict__ Wskip,
                              const float* __restrict__ bq, const float* __restrict__ bk,
                              const float* __restrict__ bv, const float* __restrict__ bskip,
                              unsigned int* __restrict__ bpack, float* __restrict__ biasp) {
    int t = blockIdx.x * blockDim.x + threadIdx.x;
    if (t < NCOLS) {
        float b;
        if      (t < 256) b = bq[t];
        else if (t < 512) b = bk[t - 256];
        else if (t < 768) b = bv[t - 512];
        else              b = bskip[t - 768];
        biasp[t] = b;
    }
    const int total = KTILES * NTILES * 32 * 8;   // 53,248
    if (t >= total) return;
    int vv   = t & 7;
    int lane = (t >> 3) & 31;
    int g    = t >> 8;
    int nt   = g % NTILES;
    int ks   = g / NTILES;
    int col  = nt * 16 + (lane & 15);
    int kbB  = (lane < 16) ? 0 : 8;
    int f    = ks * 32 + ((vv < 4) ? (kbB + 2 * vv) : (kbB + 8 + 2 * vv));
    auto fetch = [&](int ff, int cc) -> float {
        if (cc < 256) return Wq[ff * 256 + cc];
        if (cc < 512) return Wk[ff * 256 + (cc - 256)];
        if (cc < 768) return Wv[ff * 256 + (cc - 512)];
        return Wskip[ff * 64 + (cc - 768)];
    };
    union { f16 h[2]; unsigned int u; } cv;
    cv.h[0] = (f16)fetch(f, col);
    cv.h[1] = (f16)fetch(f + 1, col);
    bpack[t] = cv.u;
}

// ---------------------------------------------------------------------------
// Fused GEMM: [q|k|v|skip](+bias) = x @ [Wq|Wk|Wv|Wskip] via v_wmma_f32_16x16x32_f16
// One wave per 16x16 tile; M,N,K all exact multiples -> no guards, EXEC all-ones.
// ---------------------------------------------------------------------------
__global__ void __launch_bounds__(256)
gemm_qkvs_kernel(const unsigned int* __restrict__ apack, const unsigned int* __restrict__ bpack,
                 const float* __restrict__ biasp,
                 float* __restrict__ qb, float* __restrict__ kb,
                 float* __restrict__ vb, float* __restrict__ skipb) {
    int wave = (blockIdx.x * blockDim.x + threadIdx.x) >> 5;
    int lane = threadIdx.x & 31;
    if (wave >= MTILES * NTILES) return;          // wave-uniform
    int mt = wave / NTILES;
    int nt = wave % NTILES;

    union { v8f v; float f[8]; } acc;
    for (int i = 0; i < 8; ++i) acc.f[i] = 0.0f;

    for (int ks = 0; ks < KTILES; ++ks) {
        union { v16h v; uint4 u4[2]; } af, bf;
        const uint4* pa = (const uint4*)(apack + (((size_t)mt * KTILES + ks) * 32 + lane) * 8);
        af.u4[0] = pa[0]; af.u4[1] = pa[1];
        const uint4* pb = (const uint4*)(bpack + (((size_t)ks * NTILES + nt) * 32 + lane) * 8);
        bf.u4[0] = pb[0]; bf.u4[1] = pb[1];
        acc.v = __builtin_amdgcn_wmma_f32_16x16x32_f16(
            false, af.v, false, bf.v, (short)0, acc.v, false, false);
    }

    int col     = nt * 16 + (lane & 15);
    int rowbase = mt * 16 + ((lane < 16) ? 0 : 8);
    float bias  = biasp[col];
    float* dst; int stride; int cc;
    if      (col < 256) { dst = qb;    stride = HC;   cc = col;       }
    else if (col < 512) { dst = kb;    stride = HC;   cc = col - 256; }
    else if (col < 768) { dst = vb;    stride = HC;   cc = col - 512; }
    else                { dst = skipb; stride = C_CH; cc = col - 768; }
    for (int r = 0; r < 8; ++r)
        dst[(size_t)(rowbase + r) * stride + cc] = acc.f[r] + bias;
}

// ---------------------------------------------------------------------------
// qWe[n,h,f] = sum_c q[n,h,c] * We[f, h*C+c]  (factorization of q . e term)
// We held transposed in LDS (conflict-free: lanes vary f -> consecutive banks)
// ---------------------------------------------------------------------------
__global__ void __launch_bounds__(256)
qwe_kernel(const float* __restrict__ qb, const float* __restrict__ We,
           float* __restrict__ qwe) {
    __shared__ float sWeT[HC * F_E];              // [col][f], 32 KB
    for (int i = threadIdx.x; i < F_E * HC; i += 256) {
        int f = i / HC, col = i % HC;
        sWeT[col * F_E + f] = We[i];
    }
    __syncthreads();
    int ln = threadIdx.x >> 7;                    // 2 nodes / block
    int r  = threadIdx.x & 127;
    int h  = r >> 5;
    int f  = r & 31;
    int n  = blockIdx.x * 2 + ln;
    const float* qrow = qb + (size_t)n * HC + h * C_CH;
    const float* w    = sWeT + (h * C_CH) * F_E + f;
    float acc = 0.0f;
    for (int c = 0; c < C_CH; ++c) acc += qrow[c] * w[c * F_E];
    qwe[(size_t)n * (H_HEADS * F_E) + h * F_E + f] = acc;
}

// ---------------------------------------------------------------------------
__global__ void init_kernel(float* __restrict__ smax, float* __restrict__ ssum,
                            float* __restrict__ agg) {
    int t = blockIdx.x * blockDim.x + threadIdx.x;
    if (t < N_NODES * H_HEADS) { smax[t] = -__builtin_inff(); ssum[t] = 0.0f; }
    if (t < N_NODES * C_CH)    agg[t] = 0.0f;
}

__device__ inline void atomicMaxF(float* addr, float val) {
    if (val >= 0.0f) atomicMax((int*)addr, __float_as_int(val));
    else             atomicMin((unsigned int*)addr, __float_as_uint(val));
}

// Pass 1: score[e,h] = (q[dst,h].k[src,h] + ea[e].qWe[dst,h]) / sqrt(C); seg-max
__global__ void score_kernel(const int* __restrict__ ei, const float* __restrict__ ea,
                             const float* __restrict__ qb, const float* __restrict__ kb,
                             const float* __restrict__ qwe,
                             float* __restrict__ score, float* __restrict__ smax) {
    int t = blockIdx.x * blockDim.x + threadIdx.x;
    if (t >= N_EDGES * H_HEADS) return;
    int e = t >> 2, h = t & 3;
    int src = ei[e], dst = ei[N_EDGES + e];
    const float4* qp = (const float4*)(qb + (size_t)dst * HC + h * C_CH);
    const float4* kp = (const float4*)(kb + (size_t)src * HC + h * C_CH);
    float s = 0.0f;
    for (int i = 0; i < 16; ++i) {
        float4 a = qp[i], b = kp[i];
        s += a.x * b.x + a.y * b.y + a.z * b.z + a.w * b.w;
    }
    const float4* wp = (const float4*)(qwe + (size_t)dst * (H_HEADS * F_E) + h * F_E);
    const float4* ep = (const float4*)(ea + (size_t)e * F_E);
    for (int i = 0; i < 8; ++i) {
        float4 a = wp[i], b = ep[i];
        s += a.x * b.x + a.y * b.y + a.z * b.z + a.w * b.w;
    }
    s *= 0.125f;                                  // 1/sqrt(64)
    score[t] = s;
    atomicMaxF(&smax[dst * H_HEADS + h], s);
}

// Pass 2: ex = exp(score - max); seg-sum
__global__ void expsum_kernel(const int* __restrict__ ei, float* __restrict__ score,
                              const float* __restrict__ smax, float* __restrict__ ssum) {
    int t = blockIdx.x * blockDim.x + threadIdx.x;
    if (t >= N_EDGES * H_HEADS) return;
    int e = t >> 2, h = t & 3;
    int dst = ei[N_EDGES + e];
    float ex = __expf(score[t] - smax[dst * H_HEADS + h]);
    score[t] = ex;
    atomicAdd(&ssum[dst * H_HEADS + h], ex);
}

// Pass 3: agg[dst,c] += (1/H) * sum_h alpha[e,h]*(v[src,h,c] + (ea@We)[h,c])
// We (32 KB) in LDS, grid-strided so the fill amortizes; 4 edges per 256-thr block/iter.
__global__ void __launch_bounds__(256)
message_kernel(const int* __restrict__ ei, const float* __restrict__ ea,
               const float* __restrict__ vb, const float* __restrict__ We,
               const float* __restrict__ score, const float* __restrict__ ssum,
               float* __restrict__ agg) {
    __shared__ float sWe[F_E * HC];               // row-major [f][col], 32 KB
    __shared__ float sEa[4][F_E];
    for (int i = threadIdx.x; i < F_E * HC; i += 256) sWe[i] = We[i];
    int c   = threadIdx.x & 63;
    int sub = threadIdx.x >> 6;
    const int nchunks = N_EDGES / 4;              // 100,000 (exact)
    for (int chunk = blockIdx.x; chunk < nchunks; chunk += gridDim.x) {
        __syncthreads();
        if (threadIdx.x < 128) {
            int le = threadIdx.x >> 5, f = threadIdx.x & 31;
            sEa[le][f] = ea[(size_t)(chunk * 4 + le) * F_E + f];
        }
        __syncthreads();
        int e   = chunk * 4 + sub;
        int src = ei[e], dst = ei[N_EDGES + e];
        float al[H_HEADS];
        for (int h = 0; h < H_HEADS; ++h)
            al[h] = score[e * H_HEADS + h] / ssum[dst * H_HEADS + h];
        float eh[H_HEADS] = {0.f, 0.f, 0.f, 0.f};
        for (int f = 0; f < F_E; ++f) {
            float eav = sEa[sub][f];
            eh[0] += eav * sWe[f * HC +   0 + c];
            eh[1] += eav * sWe[f * HC +  64 + c];
            eh[2] += eav * sWe[f * HC + 128 + c];
            eh[3] += eav * sWe[f * HC + 192 + c];
        }
        const float* vrow = vb + (size_t)src * HC;
        float acc = 0.0f;
        for (int h = 0; h < H_HEADS; ++h)
            acc += al[h] * (vrow[h * C_CH + c] + eh[h]);
        atomicAdd(&agg[(size_t)dst * C_CH + c], acc * 0.25f);
    }
}

// Final: relu(agg+skip) -> BN(eval) -> @Wc + bc
__global__ void head_kernel(const float* __restrict__ agg, const float* __restrict__ skipb,
                            const float* __restrict__ gamma, const float* __restrict__ beta,
                            const float* __restrict__ rmean, const float* __restrict__ rvar,
                            const float* __restrict__ Wc, const float* __restrict__ bc,
                            float* __restrict__ out) {
    int t = blockIdx.x * blockDim.x + threadIdx.x;
    if (t >= N_NODES * NCLASS) return;
    int n = t >> 4, j = t & 15;
    float acc = bc[j];
    for (int c = 0; c < C_CH; ++c) {
        float o = agg[(size_t)n * C_CH + c] + skipb[(size_t)n * C_CH + c];
        o = fmaxf(o, 0.0f);
        o = (o - rmean[c]) * rsqrtf(rvar[c] + BN_EPS) * gamma[c] + beta[c];
        acc += o * Wc[c * NCLASS + j];
    }
    out[t] = acc;
}

// ---------------------------------------------------------------------------
extern "C" void kernel_launch(void* const* d_in, const int* in_sizes, int n_in,
                              void* d_out, int out_size, void* d_ws, size_t ws_size,
                              hipStream_t stream) {
    const float* x     = (const float*)d_in[0];
    const int*   ei    = (const int*)  d_in[1];
    const float* ea    = (const float*)d_in[2];
    const float* Wq    = (const float*)d_in[3];
    const float* bq    = (const float*)d_in[4];
    const float* Wk    = (const float*)d_in[5];
    const float* bk    = (const float*)d_in[6];
    const float* Wv    = (const float*)d_in[7];
    const float* bv    = (const float*)d_in[8];
    const float* We    = (const float*)d_in[9];
    const float* Wskip = (const float*)d_in[10];
    const float* bskip = (const float*)d_in[11];
    const float* gamma = (const float*)d_in[12];
    const float* beta  = (const float*)d_in[13];
    const float* rmean = (const float*)d_in[14];
    const float* rvar  = (const float*)d_in[15];
    const float* Wc    = (const float*)d_in[16];
    const float* bc    = (const float*)d_in[17];
    float* out = (float*)d_out;

    char* ws = (char*)d_ws;
    size_t off = 0;
    auto alloc = [&](size_t bytes) -> char* {
        char* p = ws + off;
        off += (bytes + 255) & ~(size_t)255;
        return p;
    };
    float*        qb    = (float*)alloc((size_t)N_NODES * HC * 4);
    float*        kb    = (float*)alloc((size_t)N_NODES * HC * 4);
    float*        vb    = (float*)alloc((size_t)N_NODES * HC * 4);
    float*        skipb = (float*)alloc((size_t)N_NODES * C_CH * 4);
    float*        qwe   = (float*)alloc((size_t)N_NODES * H_HEADS * F_E * 4);
    float*        smax  = (float*)alloc((size_t)N_NODES * H_HEADS * 4);
    float*        ssum  = (float*)alloc((size_t)N_NODES * H_HEADS * 4);
    float*        agg   = (float*)alloc((size_t)N_NODES * C_CH * 4);
    float*        score = (float*)alloc((size_t)N_EDGES * H_HEADS * 4);
    unsigned int* apack = (unsigned int*)alloc((size_t)MTILES * KTILES * 32 * 8 * 4);
    unsigned int* bpack = (unsigned int*)alloc((size_t)KTILES * NTILES * 32 * 8 * 4);
    float*        biasp = (float*)alloc((size_t)NCOLS * 4);

    const int TPB = 256;
    pack_a_kernel<<<(MTILES * KTILES * 32 * 8 + TPB - 1) / TPB, TPB, 0, stream>>>(x, apack);
    pack_b_kernel<<<(KTILES * NTILES * 32 * 8 + TPB - 1) / TPB, TPB, 0, stream>>>(
        Wq, Wk, Wv, Wskip, bq, bk, bv, bskip, bpack, biasp);
    init_kernel<<<(N_NODES * C_CH + TPB - 1) / TPB, TPB, 0, stream>>>(smax, ssum, agg);

    int gemm_threads = MTILES * NTILES * 32;      // one wave per 16x16 tile
    gemm_qkvs_kernel<<<(gemm_threads + TPB - 1) / TPB, TPB, 0, stream>>>(
        apack, bpack, biasp, qb, kb, vb, skipb);

    qwe_kernel<<<N_NODES / 2, TPB, 0, stream>>>(qb, We, qwe);

    score_kernel<<<(N_EDGES * H_HEADS + TPB - 1) / TPB, TPB, 0, stream>>>(
        ei, ea, qb, kb, qwe, score, smax);
    expsum_kernel<<<(N_EDGES * H_HEADS + TPB - 1) / TPB, TPB, 0, stream>>>(
        ei, score, smax, ssum);
    message_kernel<<<4096, TPB, 0, stream>>>(ei, ea, vb, We, score, ssum, agg);

    head_kernel<<<(N_NODES * NCLASS + TPB - 1) / TPB, TPB, 0, stream>>>(
        agg, skipb, gamma, beta, rmean, rvar, Wc, bc, out);
}
